// RadLIFLayer_103079215142
// MI455X (gfx1250) — matrix operand order repaired and checked by reading, not compiled, not run on hardware
//
#include <hip/hip_runtime.h>
#include <hip/hip_bf16.h>

// ---------------------------------------------------------------------------
// RadLIF layer for MI455X (gfx1250, wave32, WMMA + TDM).
//  k_init    : zero BN accumulators + grid-barrier counter
//  k_xcast   : pre-pack x into bf16, LDS-image layout [t][64][712] (zero-pad K)
//  k_pack    : pre-pack Vm^T (zero diag) [1024][1032] and Ww [1024][712] bf16,
//              initial spike buffer from s0
//  k_stats   : bf16 WMMA GEMM pass (x @ Ww^T), per-channel sum/sumsq via
//              ds_add_f32 then global_atomic_add_f32 (BN over 64000 rows)
//  k_bnfinal : BN scale/shift
//  k_scan    : persistent 16-WGP kernel, 1000 sequential steps; per step:
//              tensor_load_to_lds of x_t tile (TENSORcnt), two bf16 WMMA
//              GEMMs (x_t@Ww^T fused with s_{t-1}@Vm), register-resident
//              LIF state in WMMA C-layout, spike exchange via L2 double
//              buffer + release/acquire grid barrier.
// ---------------------------------------------------------------------------

typedef __bf16 v8bf  __attribute__((ext_vector_type(8)));
typedef __bf16 v16bf __attribute__((ext_vector_type(16)));
typedef float  v8f   __attribute__((ext_vector_type(8)));

#define B_   64
#define T_   1000
#define IN_  700
#define INP_ 704          // K padded to multiple of 32 for the x-GEMM
#define H_   1024
#define NB_  16           // persistent scan blocks (one WGP each)

// row strides in bf16 halves (padded: 16-lane row accesses hit distinct banks)
#define LDSK 40           // stats kernel A/B tiles (32 + 8)
#define LDSW 712          // 704 + 8
#define LDSV 1032         // 1024 + 8
#define XROW (B_ * LDSW)                 // 45568 halves per timestep tile
#define SMEM_SCAN ((64 * LDSV + 64 * LDSW + 64 * LDSW) * 2)   // 314368 B

// workspace layout (bytes, 16B aligned)
#define OFF_SUM    0
#define OFF_SUMSQ  4096
#define OFF_SCALE  8192
#define OFF_SHIFT  12288
#define OFF_CTR    16384
#define OFF_SB0    20480
#define OFF_SB1    (OFF_SB0 + B_ * H_ * 2)
#define OFF_VMTP   (OFF_SB1 + B_ * H_ * 2)          // 1024 x 1032 bf16
#define OFF_WWTP   (OFF_VMTP + H_ * LDSV * 2)       // 1024 x 712  bf16
#define OFF_XTP    (OFF_WWTP + H_ * LDSW * 2)       // 1000 x 64 x 712 bf16

// clip limits
#define AL_LO 0.8187307530779818f   // exp(-1/5)
#define AL_HI 0.9607894391523232f   // exp(-1/25)
#define BE_LO 0.9672161004820059f   // exp(-1/30)
#define BE_HI 0.9917012934429860f   // exp(-1/120)

__device__ __forceinline__ unsigned short f2bfu(float f) {
  unsigned u = __builtin_bit_cast(unsigned, f);
  u += 0x7FFFu + ((u >> 16) & 1u);               // round-to-nearest-even
  return (unsigned short)(u >> 16);
}

__device__ __forceinline__ v16bf cat16(v8bf lo, v8bf hi) {
  return __builtin_shufflevector(lo, hi, 0, 1, 2, 3, 4, 5, 6, 7,
                                 8, 9, 10, 11, 12, 13, 14, 15);
}

__device__ __forceinline__ v8f wmma_bf16(v16bf a, v16bf b, v8f c) {
  return __builtin_amdgcn_wmma_f32_16x16x32_bf16(false, a, false, b,
                                                 (short)0, c, false, false);
}

// ---- Tensor Data Mover: verbatim 2D bf16 tile, global -> LDS --------------
#if defined(__has_builtin)
#if __has_builtin(__builtin_amdgcn_tensor_load_to_lds) && \
    __has_builtin(__builtin_amdgcn_s_wait_tensorcnt)
#define HAVE_TDM 1
#endif
#endif

#if defined(HAVE_TDM)
typedef unsigned int tdm_v4u __attribute__((ext_vector_type(4)));
typedef int          tdm_v8i __attribute__((ext_vector_type(8)));
typedef int          tdm_v4i __attribute__((ext_vector_type(4)));

// Copy a (rows x row_halves) bf16 tile, rows contiguous at stride row_halves,
// verbatim into LDS (global buffer is pre-packed in the LDS image layout).
// 2-D tensor: only descriptor groups 0 and 1 carry state; remaining register
// groups (for 3D..5D) are zero.
__device__ __forceinline__ void tdm_load_2d(const unsigned short* gptr,
                                            void* ldsptr,
                                            unsigned row_halves, unsigned rows) {
  unsigned long long ga = (unsigned long long)gptr;
  unsigned lds = (unsigned)(unsigned long long)ldsptr;  // addr[31:0] = LDS offset
  tdm_v4u g0;
  g0[0] = 1u;                                   // count = 1 valid descriptor
  g0[1] = lds;                                  // lds_addr (bytes)
  g0[2] = (unsigned)ga;                         // global_addr[31:0]
  g0[3] = (unsigned)(ga >> 32) | 0x80000000u;   // global_addr[56:32] | type=2
  tdm_v8i g1;
  g1[0] = (int)(1u << 16);                      // data_size = 2 bytes, no pad
  g1[1] = (int)((row_halves & 0xFFFFu) << 16);  // tensor_dim0[15:0]
  g1[2] = (int)((row_halves >> 16) | ((rows & 0xFFFFu) << 16));   // dim0 hi|dim1 lo
  g1[3] = (int)((rows >> 16) | ((row_halves & 0xFFFFu) << 16));   // dim1 hi|tile0
  g1[4] = (int)(rows & 0xFFFFu);                // tile_dim1 (tile_dim2 = 0)
  g1[5] = (int)row_halves;                      // tensor_dim0_stride[31:0]
  g1[6] = 0;
  g1[7] = 0;
  tdm_v4i z4 = {0, 0, 0, 0};
  tdm_v8i z8 = {0, 0, 0, 0, 0, 0, 0, 0};
  __builtin_amdgcn_tensor_load_to_lds(g0, g1, z4, z4, z8, 0);
}
#endif

// ---------------------------------------------------------------------------
__global__ void k_init(float* gsum, float* gsumsq, unsigned* ctr) {
  int i = threadIdx.x;
  for (int c = i; c < H_; c += 256) { gsum[c] = 0.f; gsumsq[c] = 0.f; }
  if (i == 0) *ctr = 0u;
}

// ---------------------------------------------------------------------------
// x [64][1000][700] f32 -> xtp [1000][64][712] bf16 (zero-padded K)
__global__ void k_xcast(const float* __restrict__ x,
                        unsigned short* __restrict__ xtp) {
  const int t = blockIdx.x, m = blockIdx.y;
  const float* src = x + ((size_t)m * T_ + t) * IN_;
  unsigned short* dst = xtp + (size_t)t * XROW + (size_t)m * LDSW;
  for (int k = threadIdx.x; k < LDSW; k += 256)
    dst[k] = (k < IN_) ? f2bfu(src[k]) : (unsigned short)0;
}

// ---------------------------------------------------------------------------
// Vm^T (zero diagonal) -> vmtp [1024][1032] bf16 ; Ww -> wwtp [1024][712] bf16
__global__ void k_pack(const float* __restrict__ V,
                       const float* __restrict__ Ww,
                       const float* __restrict__ s0,
                       unsigned short* __restrict__ vmtp,
                       unsigned short* __restrict__ wwtp,
                       unsigned short* __restrict__ sb0) {
  const size_t tid = (size_t)blockIdx.x * blockDim.x + threadIdx.x;
  const size_t nt  = (size_t)gridDim.x * blockDim.x;
  for (size_t i = tid; i < (size_t)H_ * LDSV; i += nt) {
    size_t n = i / LDSV, k = i % LDSV;
    float v = (k < H_ && k != n) ? V[k * H_ + n] : 0.f;
    vmtp[i] = f2bfu(v);
  }
  for (size_t i = tid; i < (size_t)H_ * LDSW; i += nt) {
    size_t n = i / LDSW, k = i % LDSW;
    float v = (k < IN_) ? Ww[n * IN_ + k] : 0.f;
    wwtp[i] = f2bfu(v);
  }
  for (size_t i = tid; i < (size_t)B_ * H_; i += nt) sb0[i] = f2bfu(s0[i]);
}

// ---------------------------------------------------------------------------
// Stats pass over pre-packed bf16 operands: pure ds_load_b128 + v_wmma loop.
__global__ __launch_bounds__(256) void k_stats(const unsigned short* __restrict__ xtp,
                                               const unsigned short* __restrict__ wwtp,
                                               float* __restrict__ gsum,
                                               float* __restrict__ gsumsq) {
  __shared__ unsigned short As[64 * LDSK];
  __shared__ unsigned short Bs[64 * LDSK];
  __shared__ float csum[64];
  __shared__ float csq[64];

  const int tid = threadIdx.x;
  const int bn  = blockIdx.x;                    // N tile (0..15)
  const size_t mb = (size_t)blockIdx.y * 64;     // row base in [0, 64000)

  if (tid < 64) { csum[tid] = 0.f; csq[tid] = 0.f; }

  const int wave = tid >> 5, lane = tid & 31;
  const int l16 = lane & 15, lhi = lane >> 4;
  const int mt = wave & 3, ng = wave >> 2;
  const int arow = mt * 16 + l16;
  const int n0 = ng * 32 + l16, n1 = n0 + 16;
  const int kb = lhi * 8, ks = lhi * 16;

  const int rrow = tid >> 2, q = tid & 3;        // staging: 4 threads / row
  const unsigned short* xr = xtp + (mb + rrow) * LDSW;   // row order irrelevant
  const unsigned short* wr = wwtp + (size_t)(bn * 64 + rrow) * LDSW;

  v8f c0 = {}, c1 = {};
  for (int k0 = 0; k0 < INP_; k0 += 32) {
    __syncthreads();
    *(uint4*)&As[rrow * LDSK + q * 8] = *(const uint4*)&xr[k0 + q * 8];
    *(uint4*)&Bs[rrow * LDSK + q * 8] = *(const uint4*)&wr[k0 + q * 8];
    __syncthreads();
    v16bf a = cat16(*reinterpret_cast<const v8bf*>(&As[arow * LDSK + kb]),
                    *reinterpret_cast<const v8bf*>(&As[arow * LDSK + 16 + kb]));
    v16bf b0 = cat16(*reinterpret_cast<const v8bf*>(&Bs[n0 * LDSK + kb]),
                     *reinterpret_cast<const v8bf*>(&Bs[n0 * LDSK + 16 + kb]));
    v16bf b1 = cat16(*reinterpret_cast<const v8bf*>(&Bs[n1 * LDSK + kb]),
                     *reinterpret_cast<const v8bf*>(&Bs[n1 * LDSK + 16 + kb]));
    c0 = wmma_bf16(a, b0, c0);
    c1 = wmma_bf16(a, b1, c1);
  }

  float s1 = 0.f, s2 = 0.f, t1 = 0.f, t2 = 0.f;
#pragma unroll
  for (int r = 0; r < 8; ++r) {
    s1 += c0[r]; s2 += c0[r] * c0[r];
    t1 += c1[r]; t2 += c1[r] * c1[r];
  }
  atomicAdd(&csum[n0], s1); atomicAdd(&csq[n0], s2);   // ds_add_f32
  atomicAdd(&csum[n1], t1); atomicAdd(&csq[n1], t2);
  __syncthreads();
  if (tid < 64) {
    atomicAdd(&gsum[bn * 64 + tid], csum[tid]);        // global_atomic_add_f32
    atomicAdd(&gsumsq[bn * 64 + tid], csq[tid]);
  }
}

// ---------------------------------------------------------------------------
__global__ void k_bnfinal(const float* __restrict__ gsum,
                          const float* __restrict__ gsumsq,
                          const float* __restrict__ gamma,
                          const float* __restrict__ bnb,
                          float* __restrict__ scale,
                          float* __restrict__ shift) {
  const int c = blockIdx.x * blockDim.x + threadIdx.x;
  if (c >= H_) return;
  const float inv = 1.0f / (float)(B_ * T_);
  float mean = gsum[c] * inv;
  float var  = gsumsq[c] * inv - mean * mean;
  float sc   = gamma[c] * rsqrtf(var + 1e-5f);
  scale[c] = sc;
  shift[c] = bnb[c] - mean * sc;
}

// ---------------------------------------------------------------------------
// Persistent scan: 16 blocks x 256 threads, 1000 sequential steps.
__global__ __launch_bounds__(256, 1) void k_scan(
    const unsigned short* __restrict__ xtp,
    const unsigned short* __restrict__ wwtp,
    const unsigned short* __restrict__ vmtp,
    const float* __restrict__ alpha_p, const float* __restrict__ beta_p,
    const float* __restrict__ a_p, const float* __restrict__ b_p,
    const float* __restrict__ u0, const float* __restrict__ w0,
    const float* __restrict__ s0,
    const float* __restrict__ scale, const float* __restrict__ shift,
    unsigned short* __restrict__ sb0, unsigned short* __restrict__ sb1,
    unsigned* __restrict__ ctr, float* __restrict__ out) {
  extern __shared__ unsigned short smem[];
  unsigned short* VmTl = smem;                          // 64 x LDSV
  unsigned short* WwTl = smem + 64 * LDSV;              // 64 x LDSW
  unsigned short* Xs   = smem + 64 * LDSV + 64 * LDSW;  // 64 x LDSW

  const int tid = threadIdx.x;
  const int bn  = blockIdx.x;                   // owns H cols [bn*64, +64)
  const int wave = tid >> 5, lane = tid & 31;
  const int l16 = lane & 15, lhi = lane >> 4;
  const int mt = wave & 3, ng = wave >> 2;
  const int arow = mt * 16 + l16;
  const int n0 = ng * 32 + l16, n1 = n0 + 16;
  const int kb = lhi * 8, ks = lhi * 16;

  // ---- one-time: pin Vm^T chunk and Ww chunk in LDS (TDM or copy) ----
#if defined(HAVE_TDM)
  if (wave == 0) {
    tdm_load_2d(vmtp + (size_t)bn * 64 * LDSV, VmTl, LDSV, 64);
    tdm_load_2d(wwtp + (size_t)bn * 64 * LDSW, WwTl, LDSW, 64);
    __builtin_amdgcn_s_wait_tensorcnt(0);
  }
#else
  for (int c = tid; c < 64 * LDSV / 8; c += 256)
    *(uint4*)&VmTl[c * 8] = *(const uint4*)&vmtp[(size_t)bn * 64 * LDSV + c * 8];
  for (int c = tid; c < 64 * LDSW / 8; c += 256)
    *(uint4*)&WwTl[c * 8] = *(const uint4*)&wwtp[(size_t)bn * 64 * LDSW + c * 8];
#endif

  // ---- per-lane channel params (C-matrix layout: n = tile*16 + l16) ----
  const int ch0 = bn * 64 + n0, ch1 = bn * 64 + n1;
  const float al0 = fminf(fmaxf(alpha_p[ch0], AL_LO), AL_HI);
  const float al1 = fminf(fmaxf(alpha_p[ch1], AL_LO), AL_HI);
  const float be0 = fminf(fmaxf(beta_p[ch0], BE_LO), BE_HI);
  const float be1 = fminf(fmaxf(beta_p[ch1], BE_LO), BE_HI);
  const float aa0 = fminf(fmaxf(a_p[ch0], -1.f), 1.f);
  const float aa1 = fminf(fmaxf(a_p[ch1], -1.f), 1.f);
  const float bb0 = fminf(fmaxf(b_p[ch0], 0.f), 2.f);
  const float bb1 = fminf(fmaxf(b_p[ch1], 0.f), 2.f);
  const float sc0 = scale[ch0], sc1 = scale[ch1];
  const float sh0 = shift[ch0], sh1 = shift[ch1];

  // ---- register-resident state in WMMA C-layout ----
  float u[2][8], w[2][8], s[2][8];
#pragma unroll
  for (int r = 0; r < 8; ++r) {
    int m = mt * 16 + lhi * 8 + r;
    u[0][r] = u0[m * H_ + ch0]; u[1][r] = u0[m * H_ + ch1];
    w[0][r] = w0[m * H_ + ch0]; w[1][r] = w0[m * H_ + ch1];
    s[0][r] = s0[m * H_ + ch0]; s[1][r] = s0[m * H_ + ch1];
  }
  __syncthreads();

  unsigned short* sbuf[2] = { sb0, sb1 };

  for (int t = 0; t < T_; ++t) {
    // ---- stage x[:, t, :] tile into LDS (one TDM op / step) ----
#if defined(HAVE_TDM)
    if (wave == 0) {
      tdm_load_2d(xtp + (size_t)t * XROW, Xs, LDSW, 64);
      __builtin_amdgcn_s_wait_tensorcnt(0);
    }
#else
    for (int c = tid; c < XROW / 8; c += 256)
      *(uint4*)&Xs[c * 8] = *(const uint4*)&xtp[(size_t)t * XROW + c * 8];
#endif
    if (t + 1 < T_)
      __builtin_prefetch(xtp + (size_t)(t + 1) * XROW, 0, 1);  // global_prefetch
    __syncthreads();

    v8f wx0 = {}, wx1 = {}, rc0 = {}, rc1 = {};
    // G1: x_t @ WwChunk^T  (K = 704)
    {
      const unsigned short* A  = Xs + arow * LDSW;
      const unsigned short* B0 = WwTl + n0 * LDSW;
      const unsigned short* B1 = WwTl + n1 * LDSW;
      for (int k0 = 0; k0 < INP_; k0 += 32) {
        v16bf a = cat16(*reinterpret_cast<const v8bf*>(&A[k0 + kb]),
                        *reinterpret_cast<const v8bf*>(&A[k0 + 16 + kb]));
        v16bf b0 = cat16(*reinterpret_cast<const v8bf*>(&B0[k0 + kb]),
                         *reinterpret_cast<const v8bf*>(&B0[k0 + 16 + kb]));
        v16bf b1 = cat16(*reinterpret_cast<const v8bf*>(&B1[k0 + kb]),
                         *reinterpret_cast<const v8bf*>(&B1[k0 + 16 + kb]));
        wx0 = wmma_bf16(a, b0, wx0);
        wx1 = wmma_bf16(a, b1, wx1);
      }
    }
    // G2: s_{t-1} @ VmChunk  (K = 1024, A straight from L2 spike buffer)
    {
      const unsigned short* A  = sbuf[t & 1] + (size_t)arow * H_;
      const unsigned short* B0 = VmTl + n0 * LDSV;
      const unsigned short* B1 = VmTl + n1 * LDSV;
      for (int k0 = 0; k0 < H_; k0 += 32) {
        v16bf a = cat16(*reinterpret_cast<const v8bf*>(&A[k0 + kb]),
                        *reinterpret_cast<const v8bf*>(&A[k0 + 16 + kb]));
        v16bf b0 = cat16(*reinterpret_cast<const v8bf*>(&B0[k0 + kb]),
                         *reinterpret_cast<const v8bf*>(&B0[k0 + 16 + kb]));
        v16bf b1 = cat16(*reinterpret_cast<const v8bf*>(&B1[k0 + kb]),
                         *reinterpret_cast<const v8bf*>(&B1[k0 + 16 + kb]));
        rc0 = wmma_bf16(a, b0, rc0);
        rc1 = wmma_bf16(a, b1, rc1);
      }
    }

    // ---- RadLIF update (pure VALU on accumulator layout) + stores ----
    unsigned short* sn = sbuf[(t & 1) ^ 1];
#pragma unroll
    for (int r = 0; r < 8; ++r) {
      int m = mt * 16 + lhi * 8 + r;
      {
        float wx = wx0[r] * sc0 + sh0;
        float wn = be0 * w[0][r] + aa0 * u[0][r] + bb0 * s[0][r];
        float un = al0 * (u[0][r] - s[0][r]) + (1.f - al0) * (wx + rc0[r] - wn);
        float sv = (un > 1.0f) ? 1.f : 0.f;
        u[0][r] = un; w[0][r] = wn; s[0][r] = sv;
        out[((size_t)m * T_ + t) * H_ + ch0] = sv;
        sn[m * H_ + ch0] = (sv > 0.5f) ? (unsigned short)0x3F80 : (unsigned short)0;
      }
      {
        float wx = wx1[r] * sc1 + sh1;
        float wn = be1 * w[1][r] + aa1 * u[1][r] + bb1 * s[1][r];
        float un = al1 * (u[1][r] - s[1][r]) + (1.f - al1) * (wx + rc1[r] - wn);
        float sv = (un > 1.0f) ? 1.f : 0.f;
        u[1][r] = un; w[1][r] = wn; s[1][r] = sv;
        out[((size_t)m * T_ + t) * H_ + ch1] = sv;
        sn[m * H_ + ch1] = (sv > 0.5f) ? (unsigned short)0x3F80 : (unsigned short)0;
      }
    }

    // ---- grid-wide release/acquire barrier (monotonic counter) ----
    __syncthreads();
    if (tid == 0) {
      __hip_atomic_fetch_add(ctr, 1u, __ATOMIC_RELEASE, __HIP_MEMORY_SCOPE_AGENT);
      unsigned tgt = (unsigned)NB_ * (unsigned)(t + 1);
      while (__hip_atomic_load(ctr, __ATOMIC_ACQUIRE, __HIP_MEMORY_SCOPE_AGENT) < tgt)
        __builtin_amdgcn_s_sleep(1);
    }
    __syncthreads();
  }
}

// ---------------------------------------------------------------------------
extern "C" void kernel_launch(void* const* d_in, const int* in_sizes, int n_in,
                              void* d_out, int out_size, void* d_ws, size_t ws_size,
                              hipStream_t stream) {
  (void)in_sizes; (void)n_in; (void)out_size; (void)ws_size;
  const float* x      = (const float*)d_in[0];
  const float* Ww     = (const float*)d_in[1];
  const float* V      = (const float*)d_in[2];
  const float* alpha  = (const float*)d_in[3];
  const float* beta   = (const float*)d_in[4];
  const float* a_p    = (const float*)d_in[5];
  const float* b_p    = (const float*)d_in[6];
  const float* gamma  = (const float*)d_in[7];
  const float* bnb    = (const float*)d_in[8];
  const float* u0     = (const float*)d_in[9];
  const float* w0     = (const float*)d_in[10];
  const float* s0     = (const float*)d_in[11];

  char* ws = (char*)d_ws;
  float* gsum   = (float*)(ws + OFF_SUM);
  float* gsumsq = (float*)(ws + OFF_SUMSQ);
  float* scale  = (float*)(ws + OFF_SCALE);
  float* shift  = (float*)(ws + OFF_SHIFT);
  unsigned* ctr = (unsigned*)(ws + OFF_CTR);
  unsigned short* sb0  = (unsigned short*)(ws + OFF_SB0);
  unsigned short* sb1  = (unsigned short*)(ws + OFF_SB1);
  unsigned short* vmtp = (unsigned short*)(ws + OFF_VMTP);
  unsigned short* wwtp = (unsigned short*)(ws + OFF_WWTP);
  unsigned short* xtp  = (unsigned short*)(ws + OFF_XTP);

  k_init<<<1, 256, 0, stream>>>(gsum, gsumsq, ctr);
  k_xcast<<<dim3(T_, B_), 256, 0, stream>>>(x, xtp);
  k_pack<<<dim3(1024), 256, 0, stream>>>(V, Ww, s0, vmtp, wwtp, sb0);
  k_stats<<<dim3(16, 1000), 256, 0, stream>>>(xtp, wwtp, gsum, gsumsq);
  k_bnfinal<<<dim3(4), 256, 0, stream>>>(gsum, gsumsq, gamma, bnb, scale, shift);
  (void)hipFuncSetAttribute((const void*)k_scan,
                            hipFuncAttributeMaxDynamicSharedMemorySize,
                            SMEM_SCAN);
  k_scan<<<NB_, 256, SMEM_SCAN, stream>>>(xtp, wwtp, vmtp, alpha, beta, a_p, b_p,
                                          u0, w0, s0, scale, shift,
                                          sb0, sb1, ctr, (float*)d_out);
}
